// BottleNeckMTB_70050916598043
// MI455X (gfx1250) — matrix operand order repaired and checked by reading, not compiled, not run on hardware
//
#include <hip/hip_runtime.h>
#include <hip/hip_bf16.h>

// ---------------------------------------------------------------------------
// Types for CDNA5 WMMA
// ---------------------------------------------------------------------------
typedef __attribute__((ext_vector_type(16))) __bf16        v16bf;
typedef __attribute__((ext_vector_type(8)))  float         v8f;
typedef __attribute__((ext_vector_type(2)))  __bf16        bf16x2;
typedef __attribute__((ext_vector_type(8)))  unsigned int  u32x8;

__device__ inline unsigned int pkbf(float a, float b) {
    bf16x2 t;
    t.x = (__bf16)a;
    t.y = (__bf16)b;
    return __builtin_bit_cast(unsigned int, t);
}

__device__ inline float gelu_tanh(float x) {
    const float c0 = 0.7978845608028654f;   // sqrt(2/pi)
    const float c1 = 0.044715f;
    float x3 = x * x * x;
    return 0.5f * x * (1.0f + tanhf(c0 * (x + c1 * x3)));
}

// ---------------------------------------------------------------------------
// Generic bf16-WMMA GEMM:  C = epilogue( scale * (A @ B) )
//   A: M x K   row-major, row stride lda, batch offset zb*sAb + zh*sAh
//   B: transB=0 -> K x N row-major (ldb);  transB=1 -> stored N x K (ldb)
//   C: M x N   row-major, ldc, batch offset zb*sCb + zh*sCh
//   z = blockIdx.z ; zb = z / hdiv ; zh = z % hdiv
//   ep bit0: +bias[col]   bit1: gelu   bit2: +res1[cidx]
//   ep bit3: *= mask[row]; += res2[cidx]   (transformer-block final)
//
// Double-buffered LDS ping-pong with register prefetch of the next K-tile,
// one barrier per K-step, and an unguarded fast path for fully in-bounds
// 128x128 blocks so the 8 staging global_load_b128 issue as a clause.
// ---------------------------------------------------------------------------
#define BLK_M 128
#define BLK_N 128
#define BLK_K 32
#define LDS_K 40   // padded stride (halves), even, 80B per row

__global__ __launch_bounds__(256)
void wmma_gemm_kernel(const float* __restrict__ A, const float* __restrict__ B,
                      const float* __restrict__ bias,
                      const float* res1, const float* res2,
                      const unsigned char* __restrict__ maskrow,
                      float* C,
                      int M, int N, int K,
                      int lda, int ldb, int ldc,
                      long sAb, long sAh, long sBb, long sBh, long sCb, long sCh,
                      int hdiv, int transB, float scale, int ep)
{
    __shared__ __bf16 As[2][BLK_M * LDS_K];
    __shared__ __bf16 Bs[2][BLK_N * LDS_K];

    const int tid  = threadIdx.x;
    const int lane = tid & 31;
    const int wid  = tid >> 5;
    const int wm   = wid >> 1;          // 0..3 : 32-row slab
    const int wn   = wid & 1;           // 0..1 : 64-col slab
    const int m0   = blockIdx.y * BLK_M;
    const int n0   = blockIdx.x * BLK_N;
    const int z    = blockIdx.z;
    const int zb   = z / hdiv;
    const int zh   = z - zb * hdiv;

    const float* Ab = A + (long)zb * sAb + (long)zh * sAh;
    const float* Bb = B + (long)zb * sBb + (long)zh * sBh;
    const long  coff = (long)zb * sCb + (long)zh * sCh;

    const int hl  = lane >> 4;          // 0/1 half-wave
    const int l15 = lane & 15;

    const bool full = (m0 + BLK_M <= M) && (n0 + BLK_N <= N);

    // per-thread staging slots
    const int a_row = tid >> 1;                 // 0..127
    const int a_kq0 = (tid & 1) << 4;           // 0 or 16 (2 float4 each, stride 4)
    // A: thread covers rows a_row, float4 at k = a_kq0 + {0,4}? -> use flat idx scheme
    float4 fa[4], fb[4];

    // ---- register prefetch of one K-tile ------------------------------------
    auto load_regs = [&](int kt) {
        const int k0 = kt * BLK_K;
        if (full) {
#pragma unroll
            for (int it = 0; it < 4; ++it) {
                int idx = tid + it * 256;       // 0..1023 float4 slots
                int row = idx >> 3;
                int kq  = (idx & 7) << 2;
                fa[it] = *(const float4*)(Ab + (long)(m0 + row) * lda + k0 + kq);
            }
            if (transB) {
#pragma unroll
                for (int it = 0; it < 4; ++it) {
                    int idx = tid + it * 256;
                    int n  = idx >> 3;
                    int kq = (idx & 7) << 2;
                    fb[it] = *(const float4*)(Bb + (long)(n0 + n) * ldb + k0 + kq);
                }
            } else {
#pragma unroll
                for (int it = 0; it < 4; ++it) {
                    int idx = tid + it * 256;
                    int k  = idx >> 5;          // 0..31
                    int nq = (idx & 31) << 2;   // 0..124
                    fb[it] = *(const float4*)(Bb + (long)(k0 + k) * ldb + n0 + nq);
                }
            }
        } else {
#pragma unroll
            for (int it = 0; it < 4; ++it) {
                int idx = tid + it * 256;
                int row = idx >> 3;
                int kq  = (idx & 7) << 2;
                int gr  = m0 + row;
                fa[it] = make_float4(0.f, 0.f, 0.f, 0.f);
                if (gr < M) fa[it] = *(const float4*)(Ab + (long)gr * lda + k0 + kq);
            }
            if (transB) {
#pragma unroll
                for (int it = 0; it < 4; ++it) {
                    int idx = tid + it * 256;
                    int n  = idx >> 3;
                    int kq = (idx & 7) << 2;
                    int gn = n0 + n;
                    fb[it] = make_float4(0.f, 0.f, 0.f, 0.f);
                    if (gn < N) fb[it] = *(const float4*)(Bb + (long)gn * ldb + k0 + kq);
                }
            } else {
#pragma unroll
                for (int it = 0; it < 4; ++it) {
                    int idx = tid + it * 256;
                    int k  = idx >> 5;
                    int nq = (idx & 31) << 2;
                    int gn = n0 + nq;
                    fb[it] = make_float4(0.f, 0.f, 0.f, 0.f);
                    if (gn < N) fb[it] = *(const float4*)(Bb + (long)(k0 + k) * ldb + gn);
                }
            }
        }
    };

    // ---- convert + store staged registers into LDS buffer -------------------
    auto store_lds = [&](int buf) {
#pragma unroll
        for (int it = 0; it < 4; ++it) {
            int idx = tid + it * 256;
            int row = idx >> 3;
            int kq  = (idx & 7) << 2;
            uint2 p;
            p.x = pkbf(fa[it].x, fa[it].y);
            p.y = pkbf(fa[it].z, fa[it].w);
            *(uint2*)(&As[buf][row * LDS_K + kq]) = p;
        }
        if (transB) {
#pragma unroll
            for (int it = 0; it < 4; ++it) {
                int idx = tid + it * 256;
                int n  = idx >> 3;
                int kq = (idx & 7) << 2;
                uint2 p;
                p.x = pkbf(fb[it].x, fb[it].y);
                p.y = pkbf(fb[it].z, fb[it].w);
                *(uint2*)(&Bs[buf][n * LDS_K + kq]) = p;
            }
        } else {
#pragma unroll
            for (int it = 0; it < 4; ++it) {
                int idx = tid + it * 256;
                int k  = idx >> 5;
                int nq = (idx & 31) << 2;
                Bs[buf][(nq + 0) * LDS_K + k] = (__bf16)fb[it].x;
                Bs[buf][(nq + 1) * LDS_K + k] = (__bf16)fb[it].y;
                Bs[buf][(nq + 2) * LDS_K + k] = (__bf16)fb[it].z;
                Bs[buf][(nq + 3) * LDS_K + k] = (__bf16)fb[it].w;
            }
        }
    };

    v8f acc[2][4];
#pragma unroll
    for (int i = 0; i < 2; ++i)
#pragma unroll
        for (int j = 0; j < 4; ++j)
#pragma unroll
            for (int r = 0; r < 8; ++r) acc[i][j][r] = 0.0f;

    // ---- fragment load + 8 WMMAs on one LDS buffer --------------------------
    auto compute = [&](int buf) {
        const unsigned int* asu = (const unsigned int*)As[buf];
        const unsigned int* bsu = (const unsigned int*)Bs[buf];
        v16bf af[2], bfr[4];
#pragma unroll
        for (int tm = 0; tm < 2; ++tm) {
            int row = wm * 32 + tm * 16 + l15;
            const unsigned int* p = asu + row * (LDS_K / 2) + hl * 4; // K = hl*8 ..
            u32x8 w;
            w[0] = p[0]; w[1] = p[1]; w[2] = p[2]; w[3] = p[3];       // K 0..7 (+off)
            w[4] = p[8]; w[5] = p[9]; w[6] = p[10]; w[7] = p[11];     // K 16..23 (+off)
            af[tm] = __builtin_bit_cast(v16bf, w);
        }
#pragma unroll
        for (int tn = 0; tn < 4; ++tn) {
            int col = wn * 64 + tn * 16 + l15;
            const unsigned int* q = bsu + col * (LDS_K / 2) + hl * 4; // K = hl*8 .., 16+hl*8 ..
            u32x8 w;
#pragma unroll
            for (int j = 0; j < 8; ++j) w[j] = q[j + (j & 4)];
            bfr[tn] = __builtin_bit_cast(v16bf, w);
        }
#pragma unroll
        for (int tm = 0; tm < 2; ++tm)
#pragma unroll
            for (int tn = 0; tn < 4; ++tn)
                acc[tm][tn] = __builtin_amdgcn_wmma_f32_16x16x32_bf16(
                    false, af[tm], false, bfr[tn], (short)0, acc[tm][tn],
                    false, false);
    };

    // ---- pipelined main loop ------------------------------------------------
    const int nk = K / BLK_K;
    load_regs(0);
    store_lds(0);
    __syncthreads();
    for (int kt = 0; kt < nk; ++kt) {
        const int cur = kt & 1;
        if (kt + 1 < nk) load_regs(kt + 1);     // HBM prefetch overlaps WMMAs below
        compute(cur);
        if (kt + 1 < nk) {
            store_lds(cur ^ 1);
            __syncthreads();
        }
    }

    // ---- epilogue -----------------------------------------------------------
#pragma unroll
    for (int tm = 0; tm < 2; ++tm) {
#pragma unroll
        for (int tn = 0; tn < 4; ++tn) {
#pragma unroll
            for (int r = 0; r < 8; ++r) {
                int gr = m0 + wm * 32 + tm * 16 + r + (hl << 3);
                int gc = n0 + wn * 64 + tn * 16 + l15;
                if (gr < M && gc < N) {
                    float v = acc[tm][tn][r] * scale;
                    long cidx = coff + (long)gr * ldc + gc;
                    if (ep & 1) v += bias[gc];
                    if (ep & 4) v += res1[cidx];
                    if (ep & 2) v = gelu_tanh(v);
                    if (ep & 8) { v *= (float)maskrow[gr]; v += res2[cidx]; }
                    C[cidx] = v;
                }
            }
        }
    }
}

// ---------------------------------------------------------------------------
// LayerNorm over last dim (one workgroup per row)
// ---------------------------------------------------------------------------
__global__ __launch_bounds__(256)
void layernorm_kernel(const float* __restrict__ x, const float* __restrict__ g,
                      const float* __restrict__ b, float* __restrict__ y, int C)
{
    __shared__ float r1[256], r2[256];
    long row = blockIdx.x;
    const float* xr = x + row * C;
    float ls = 0.f, lq = 0.f;
    for (int c = threadIdx.x; c < C; c += 256) { float v = xr[c]; ls += v; lq += v * v; }
    r1[threadIdx.x] = ls; r2[threadIdx.x] = lq;
    __syncthreads();
    for (int s = 128; s > 0; s >>= 1) {
        if ((int)threadIdx.x < s) { r1[threadIdx.x] += r1[threadIdx.x + s]; r2[threadIdx.x] += r2[threadIdx.x + s]; }
        __syncthreads();
    }
    float mean = r1[0] / C;
    float var  = r2[0] / C - mean * mean;
    float rs   = rsqrtf(var + 1e-5f);
    float* yr = y + row * C;
    for (int c = threadIdx.x; c < C; c += 256)
        yr[c] = (xr[c] - mean) * rs * g[c] + b[c];
}

// ---------------------------------------------------------------------------
// Row softmax in place (optional key mask -> -1e9)
// ---------------------------------------------------------------------------
__global__ __launch_bounds__(256)
void softmax_kernel(float* __restrict__ s, int cols,
                    const unsigned char* __restrict__ mask,
                    int rowsPerBatch, int hdiv, int Tmask)
{
    __shared__ float red[256];
    long row = blockIdx.x;
    float* sr = s + row * (long)cols;
    const unsigned char* mb = nullptr;
    if (mask) { long bidx = (row / rowsPerBatch) / hdiv; mb = mask + bidx * (long)Tmask; }

    float lmax = -3.4e38f;
    for (int c = threadIdx.x; c < cols; c += 256) {
        float v = sr[c];
        if (mb && !mb[c]) v = -1e9f;
        lmax = fmaxf(lmax, v);
    }
    red[threadIdx.x] = lmax; __syncthreads();
    for (int t = 128; t > 0; t >>= 1) {
        if ((int)threadIdx.x < t) red[threadIdx.x] = fmaxf(red[threadIdx.x], red[threadIdx.x + t]);
        __syncthreads();
    }
    float mx = red[0];
    __syncthreads();

    float lsum = 0.f;
    for (int c = threadIdx.x; c < cols; c += 256) {
        float v = sr[c];
        if (mb && !mb[c]) v = -1e9f;
        float e = __expf(v - mx);
        sr[c] = e; lsum += e;
    }
    red[threadIdx.x] = lsum; __syncthreads();
    for (int t = 128; t > 0; t >>= 1) {
        if ((int)threadIdx.x < t) red[threadIdx.x] += red[threadIdx.x + t];
        __syncthreads();
    }
    float inv = 1.0f / red[0];
    for (int c = threadIdx.x; c < cols; c += 256) sr[c] *= inv;
}

// ---------------------------------------------------------------------------
// Row-wise concat along sequence: dst[b] = [first[b] (T rows); second[b] (T rows)]
// ---------------------------------------------------------------------------
__global__ __launch_bounds__(256)
void concat_rows_kernel(const float4* __restrict__ first, const float4* __restrict__ second,
                        float4* __restrict__ dst, int Trows, int C4, long total4)
{
    long idx = (long)blockIdx.x * 256 + threadIdx.x;
    if (idx >= total4) return;
    long per_b = (long)2 * Trows * C4;
    long b   = idx / per_b;
    long rem = idx - b * per_b;
    long r   = rem / C4;
    long c   = rem - r * C4;
    float4 v = (r < Trows) ? first[(b * Trows + r) * C4 + c]
                           : second[(b * Trows + (r - Trows)) * C4 + c];
    dst[idx] = v;
}

// ---------------------------------------------------------------------------
// Host orchestration
// ---------------------------------------------------------------------------
static void gemm(hipStream_t st, const float* A, const float* B, const float* bias,
                 const float* res1, const float* res2, const unsigned char* mask,
                 float* C, int M, int N, int K, int lda, int ldb, int ldc,
                 long sAb, long sAh, long sBb, long sBh, long sCb, long sCh,
                 int batches, int hdiv, int transB, float scale, int ep)
{
    dim3 g((N + BLK_N - 1) / BLK_N, (M + BLK_M - 1) / BLK_M, batches);
    wmma_gemm_kernel<<<g, 256, 0, st>>>(A, B, bias, res1, res2, mask, C,
                                        M, N, K, lda, ldb, ldc,
                                        sAb, sAh, sBb, sBh, sCb, sCh,
                                        hdiv, transB, scale, ep);
}

// model constants
static const long NB = 16, NT = 512, NC = 512, NCA = 128, NL = 4, NH = 4, NBN = 64, NDH = 128, NC4H = 2048;
static const long SZ_BT_C = NB * NT * NC;           // 4,194,304 floats

static void run_block(hipStream_t st, const float* x, const float* const p[16], int i,
                      float* stream_buf, const unsigned char* masks,
                      float* h_buf, float* q_buf, float* k_buf, float* vv_buf,
                      float* big, float* o_buf, float* x1_buf)
{
    const long C = NC, T = NT, Bn = NB, H = NH, DH = NDH, C4 = NC4H;
    const float sc_d = 0.08838834764831845f;        // 128^-0.5

    layernorm_kernel<<<(unsigned)(Bn * T), 256, 0, st>>>(x, p[0] + i * C, p[1] + i * C, h_buf, (int)C);
    // q, k, v projections
    gemm(st, h_buf, p[2] + (long)i * C * C, p[3] + i * C, nullptr, nullptr, nullptr, q_buf,
         (int)(Bn * T), (int)C, (int)C, (int)C, (int)C, (int)C, 0, 0, 0, 0, 0, 0, 1, 1, 0, 1.0f, 1);
    gemm(st, h_buf, p[4] + (long)i * C * C, p[5] + i * C, nullptr, nullptr, nullptr, k_buf,
         (int)(Bn * T), (int)C, (int)C, (int)C, (int)C, (int)C, 0, 0, 0, 0, 0, 0, 1, 1, 0, 1.0f, 1);
    gemm(st, h_buf, p[6] + (long)i * C * C, p[7] + i * C, nullptr, nullptr, nullptr, vv_buf,
         (int)(Bn * T), (int)C, (int)C, (int)C, (int)C, (int)C, 0, 0, 0, 0, 0, 0, 1, 1, 0, 1.0f, 1);
    // scores = Q @ K^T per (b, h)
    gemm(st, q_buf, k_buf, nullptr, nullptr, nullptr, nullptr, big,
         (int)T, (int)T, (int)DH, (int)C, (int)C, (int)T,
         T * C, DH, T * C, DH, H * T * T, T * T,
         (int)(Bn * H), (int)H, 1, sc_d, 0);
    softmax_kernel<<<(unsigned)(Bn * H * T), 256, 0, st>>>(big, (int)T, masks, (int)T, (int)H, (int)T);
    // o = P @ V per (b, h), scattered into head columns
    gemm(st, big, vv_buf, nullptr, nullptr, nullptr, nullptr, o_buf,
         (int)T, (int)DH, (int)T, (int)T, (int)C, (int)C,
         H * T * T, T * T, T * C, DH, T * C, DH,
         (int)(Bn * H), (int)H, 0, 1.0f, 0);
    // x1 = x + o @ wo + bo
    gemm(st, o_buf, p[8] + (long)i * C * C, p[9] + i * C, x, nullptr, nullptr, x1_buf,
         (int)(Bn * T), (int)C, (int)C, (int)C, (int)C, (int)C, 0, 0, 0, 0, 0, 0, 1, 1, 0, 1.0f, 1 | 4);
    // MLP
    layernorm_kernel<<<(unsigned)(Bn * T), 256, 0, st>>>(x1_buf, p[10] + i * C, p[11] + i * C, h_buf, (int)C);
    gemm(st, h_buf, p[12] + (long)i * C * C4, p[13] + i * C4, nullptr, nullptr, nullptr, big,
         (int)(Bn * T), (int)C4, (int)C, (int)C, (int)C4, (int)C4, 0, 0, 0, 0, 0, 0, 1, 1, 0, 1.0f, 1 | 2);
    // stream = (x1 + big @ w2 + b2) * mask + stream_old
    gemm(st, big, p[14] + (long)i * C4 * C, p[15] + i * C, x1_buf, stream_buf, masks, stream_buf,
         (int)(Bn * T), (int)C, (int)C4, (int)C4, (int)C, (int)C, 0, 0, 0, 0, 0, 0, 1, 1, 0, 1.0f, 1 | 4 | 8);
}

extern "C" void kernel_launch(void* const* d_in, const int* in_sizes, int n_in,
                              void* d_out, int out_size, void* d_ws, size_t ws_size,
                              hipStream_t stream)
{
    (void)in_sizes; (void)n_in; (void)out_size; (void)ws_size;

    const float* out_video  = (const float*)d_in[0];
    const float* out_audio  = (const float*)d_in[1];
    const unsigned char* masks = (const unsigned char*)d_in[2];   // (B,T,1) bool
    const float* w_down     = (const float*)d_in[3];
    const float* b_down     = (const float*)d_in[4];
    const float* bottlenecks = (const float*)d_in[5];
    const float* pv[16];
    const float* pa[16];
    for (int j = 0; j < 16; ++j) {
        pv[j] = (const float*)d_in[6 + j];
        pa[j] = (const float*)d_in[22 + j];
    }

    // ---- workspace layout (floats) ----
    float* ws = (float*)d_ws;
    long off = 0;
    float* a_buf  = ws + off; off += SZ_BT_C;
    float* v_buf  = ws + off; off += SZ_BT_C;
    float* kv_buf = ws + off; off += 2 * SZ_BT_C;
    float* s1     = ws + off; off += NB * NBN * 2 * NT;   // (B,64,1024)
    float* fused  = ws + off; off += NB * NBN * NC;       // (B,64,512)
    float* s2     = ws + off; off += NB * NT * NBN;       // (B,512,64)
    float* f_v    = ws + off; off += SZ_BT_C;
    float* f_a    = ws + off; off += SZ_BT_C;
    float* h_buf  = ws + off; off += SZ_BT_C;
    float* q_buf  = ws + off; off += SZ_BT_C;
    float* k_buf  = ws + off; off += SZ_BT_C;
    float* vv_buf = ws + off; off += SZ_BT_C;
    float* big    = ws + off; off += NB * NH * NT * NT;   // scores / MLP hidden (64 MB)
    float* o_buf  = ws + off; off += SZ_BT_C;
    float* x1_buf = ws + off; off += SZ_BT_C;

    const float sc_c = 0.044194173824159216f;   // 512^-0.5
    const long  TOT4 = NB * 2 * NT * (NC / 4);  // float4 count for (B,2T,C)
    const unsigned CCG = (unsigned)((TOT4 + 255) / 256);

    // v = out_video ; a = out_audio @ w_down + b_down
    hipMemcpyAsync(v_buf, out_video, SZ_BT_C * sizeof(float), hipMemcpyDeviceToDevice, stream);
    gemm(stream, out_audio, w_down, b_down, nullptr, nullptr, nullptr, a_buf,
         (int)(NB * NT), (int)NC, (int)NCA, (int)NCA, (int)NC, (int)NC,
         0, 0, 0, 0, 0, 0, 1, 1, 0, 1.0f, 1);

    for (int i = 0; i < (int)NL; ++i) {
        const float* lat = bottlenecks + (long)i * NBN * NC;

        // kv = concat([a, v], seq)
        concat_rows_kernel<<<CCG, 256, 0, stream>>>((const float4*)a_buf, (const float4*)v_buf,
                                                    (float4*)kv_buf, (int)NT, (int)(NC / 4), TOT4);
        // S1 = lat @ kv^T * C^-0.5 ; softmax ; fused = P1 @ kv
        gemm(stream, lat, kv_buf, nullptr, nullptr, nullptr, nullptr, s1,
             (int)NBN, (int)(2 * NT), (int)NC, (int)NC, (int)NC, (int)(2 * NT),
             0, 0, 2 * NT * NC, 0, NBN * 2 * NT, 0, (int)NB, 1, 1, sc_c, 0);
        softmax_kernel<<<(unsigned)(NB * NBN), 256, 0, stream>>>(s1, (int)(2 * NT), nullptr, 1, 1, 1);
        gemm(stream, s1, kv_buf, nullptr, nullptr, nullptr, nullptr, fused,
             (int)NBN, (int)NC, (int)(2 * NT), (int)(2 * NT), (int)NC, (int)NC,
             NBN * 2 * NT, 0, 2 * NT * NC, 0, NBN * NC, 0, (int)NB, 1, 0, 1.0f, 0);

        // f_a = a + attn(a, fused, fused)
        gemm(stream, a_buf, fused, nullptr, nullptr, nullptr, nullptr, s2,
             (int)NT, (int)NBN, (int)NC, (int)NC, (int)NC, (int)NBN,
             NT * NC, 0, NBN * NC, 0, NT * NBN, 0, (int)NB, 1, 1, sc_c, 0);
        softmax_kernel<<<(unsigned)(NB * NT), 256, 0, stream>>>(s2, (int)NBN, nullptr, 1, 1, 1);
        gemm(stream, s2, fused, nullptr, a_buf, nullptr, nullptr, f_a,
             (int)NT, (int)NC, (int)NBN, (int)NBN, (int)NC, (int)NC,
             NT * NBN, 0, NBN * NC, 0, NT * NC, 0, (int)NB, 1, 0, 1.0f, 4);

        // f_v = v + attn(v, fused, fused)
        gemm(stream, v_buf, fused, nullptr, nullptr, nullptr, nullptr, s2,
             (int)NT, (int)NBN, (int)NC, (int)NC, (int)NC, (int)NBN,
             NT * NC, 0, NBN * NC, 0, NT * NBN, 0, (int)NB, 1, 1, sc_c, 0);
        softmax_kernel<<<(unsigned)(NB * NT), 256, 0, stream>>>(s2, (int)NBN, nullptr, 1, 1, 1);
        gemm(stream, s2, fused, nullptr, v_buf, nullptr, nullptr, f_v,
             (int)NT, (int)NC, (int)NBN, (int)NBN, (int)NC, (int)NC,
             NT * NBN, 0, NBN * NC, 0, NT * NC, 0, (int)NB, 1, 0, 1.0f, 4);

        // v = block_v(f_v) + v ;  a = block_a(f_a) + a
        run_block(stream, f_v, pv, i, v_buf, masks, h_buf, q_buf, k_buf, vv_buf, big, o_buf, x1_buf);
        run_block(stream, f_a, pa, i, a_buf, masks, h_buf, q_buf, k_buf, vv_buf, big, o_buf, x1_buf);
    }

    // out = concat([v, a], seq)
    concat_rows_kernel<<<CCG, 256, 0, stream>>>((const float4*)v_buf, (const float4*)a_buf,
                                                (float4*)d_out, (int)NT, (int)(NC / 4), TOT4);
}
